// GumbelVQ_11879879544401
// MI455X (gfx1250) — matrix operand order repaired and checked by reading, not compiled, not run on hardware
//
#include <hip/hip_runtime.h>
#include <hip/hip_bf16.h>

typedef __attribute__((ext_vector_type(16))) __bf16 v16bf;
typedef __attribute__((ext_vector_type(8)))  __bf16 v8bf;
typedef __attribute__((ext_vector_type(8)))  float  v8f;

#define D_DIM   256
#define K_CB    8192
#define N_PIX   16384      // 16*32*32
#define ROWS    16         // pixel rows per block
#define NTILES  (K_CB / 16)   // 512 column tiles in GEMM1
#define NWAVES  8          // 256 threads = 8 wave32

// ---------------------------------------------------------------------------
// Prep: codebook f32 -> bf16 (row major + transposed) and c2[k] = ||c_k||^2
// ---------------------------------------------------------------------------
__global__ void __launch_bounds__(256) vq_prep(const float* __restrict__ cb,
                                               __bf16* __restrict__ cb16,
                                               __bf16* __restrict__ cbT16,
                                               float* __restrict__ c2) {
    int k = blockIdx.x;          // 0..8191
    int d = threadIdx.x;         // 0..255
    float v = cb[(size_t)k * D_DIM + d];
    cb16[(size_t)k * D_DIM + d] = (__bf16)v;
    cbT16[(size_t)d * K_CB + k] = (__bf16)v;
    __shared__ float red[256];
    red[d] = v * v;
    __syncthreads();
    for (int s = 128; s > 0; s >>= 1) {
        if (d < s) red[d] += red[d + s];
        __syncthreads();
    }
    if (d == 0) c2[k] = red[0];
}

// ---------------------------------------------------------------------------
// Fused main kernel: distances (WMMA) -> logits -> softmax/argmin in LDS ->
// encodings write -> quantized = probs @ codebook (WMMA)
// LDS: 16x8192 bf16 probs tile (256KB) + 16x256 bf16 x tile + reductions
// ---------------------------------------------------------------------------
#define SMEM_PROBS   (ROWS * K_CB * 2)        // 262144 B
#define SMEM_XA      (ROWS * D_DIM * 2)       // 8192 B
#define SMEM_RED     (4 * (16 + 128 + 128 + 128 + 16 + 16 + 16 + 16))
#define SMEM_BYTES   (SMEM_PROBS + SMEM_XA + SMEM_RED)

__global__ void __launch_bounds__(256, 1)
vq_main(const float* __restrict__ x,
        const float* __restrict__ noise,
        const __bf16* __restrict__ cb16,
        const __bf16* __restrict__ cbT16,
        const float*  __restrict__ c2,
        float* __restrict__ q_out,      // [N_PIX][256]
        float* __restrict__ enc_out,    // [N_PIX][8192]
        int*   __restrict__ idx_out) {  // [N_PIX]
    extern __shared__ char smem[];
    __bf16* sP   = (__bf16*)smem;                       // [16][8192] logits/probs
    __bf16* sA   = (__bf16*)(smem + SMEM_PROBS);        // [16][256] x tile
    float*  sX2  = (float*)(smem + SMEM_PROBS + SMEM_XA); // [16]
    float*  sWMax = sX2 + 16;                           // [8][16]
    float*  sWMin = sWMax + 128;                        // [8][16]
    int*    sWIdx = (int*)(sWMin + 128);                // [8][16]
    float*  sRowMax = (float*)(sWIdx + 128);            // [16]
    float*  sRowSum = sRowMax + 16;                     // [16]
    float*  sRowInv = sRowSum + 16;                     // [16]
    int*    sRowIdx = (int*)(sRowInv + 16);             // [16]

    const int t    = threadIdx.x;
    const int lane = t & 31;
    const int wave = t >> 5;
    const int ncol = lane & 15;          // N column within a 16x16 C tile
    const int hi8  = (lane >> 4) * 8;    // row offset for C layout (lanes 16-31 -> M+8)
    const size_t pix0 = (size_t)blockIdx.x * ROWS;

    // ---------------- Phase 1: stage x tile (bf16) + x2 row norms ----------
    if (t < 16) { sX2[t] = 0.f; sRowSum[t] = 0.f; }
    __syncthreads();
    for (int i = t; i < ROWS * D_DIM; i += 256) {
        int m = i >> 8, d = i & 255;
        float v = x[(pix0 + m) * D_DIM + d];
        sA[m * D_DIM + d] = (__bf16)v;
        atomicAdd(&sX2[m], v * v);       // ds_add_f32
    }
    __syncthreads();

    // ---------------- Phase 2: GEMM1 + fused logits/argmax/argmin ----------
    // Preload A fragments for all 8 k-steps of D=256 from LDS.
    // A k order used here (B matches): lane<16 -> K {0..15}; lane>=16 -> K {16..31}
    v16bf afrag[8];
    {
        const int m  = lane & 15;
        const int bo = (lane >> 4) ? 16 : 0;
        #pragma unroll
        for (int ks = 0; ks < 8; ++ks) {
            const __bf16* ap = &sA[m * D_DIM + ks * 32 + bo];
            v8bf lo = *(const v8bf*)(ap);
            v8bf hi = *(const v8bf*)(ap + 8);
            v16bf a;
            #pragma unroll
            for (int i = 0; i < 8; ++i) { a[i] = lo[i]; a[i + 8] = hi[i]; }
            afrag[ks] = a;
        }
    }
    float x2r[8];
    #pragma unroll
    for (int r = 0; r < 8; ++r) x2r[r] = sX2[r + hi8];

    float rmax[8], rmind[8]; int rmidx[8];
    #pragma unroll
    for (int r = 0; r < 8; ++r) { rmax[r] = -3.0e38f; rmind[r] = 3.0e38f; rmidx[r] = 0; }

    for (int tile = wave; tile < NTILES; tile += NWAVES) {
        const int n0 = tile * 16;
        const int n  = n0 + ncol;
        const __bf16* brow = &cb16[(size_t)n * D_DIM + ((lane >> 4) << 4)];
        __builtin_prefetch(&cb16[(size_t)(n + NWAVES * 16) * D_DIM], 0, 0);
        v8f acc = {};
        #pragma unroll
        for (int ks = 0; ks < 8; ++ks) {
            // B 32x16 layout: lane holds column n, 16 contiguous K values
            v16bf b = *(const v16bf*)(brow + ks * 32);
            acc = __builtin_amdgcn_wmma_f32_16x16x32_bf16(
                      false, afrag[ks], false, b, (short)0, acc, false, false);
        }
        const float c2n = c2[n];
        #pragma unroll
        for (int r = 0; r < 8; ++r) {
            const int m = r + hi8;
            float dist = x2r[r] - 2.0f * acc[r] + c2n;
            float lg   = noise[(pix0 + m) * (size_t)K_CB + n] - dist;  // TAU = 1
            if (dist < rmind[r]) { rmind[r] = dist; rmidx[r] = n; }
            rmax[r] = fmaxf(rmax[r], lg);
            sP[m * K_CB + n] = (__bf16)lg;
        }
    }

    // half-wave (16-lane) reductions: lanes 0-15 share row r, 16-31 share r+8
    #pragma unroll
    for (int r = 0; r < 8; ++r) {
        float mx = rmax[r], mn = rmind[r]; int mi = rmidx[r];
        #pragma unroll
        for (int off = 1; off < 16; off <<= 1) {
            mx = fmaxf(mx, __shfl_xor(mx, off, 32));
            float omn = __shfl_xor(mn, off, 32);
            int   omi = __shfl_xor(mi, off, 32);
            if (omn < mn) { mn = omn; mi = omi; }
        }
        if (ncol == 0) {
            const int m = r + hi8;
            sWMax[wave * 16 + m] = mx;
            sWMin[wave * 16 + m] = mn;
            sWIdx[wave * 16 + m] = mi;
        }
    }
    __syncthreads();
    if (t < 16) {
        float mx = -3.0e38f, mn = 3.0e38f; int mi = 0;
        #pragma unroll
        for (int w = 0; w < NWAVES; ++w) {
            mx = fmaxf(mx, sWMax[w * 16 + t]);
            if (sWMin[w * 16 + t] < mn) { mn = sWMin[w * 16 + t]; mi = sWIdx[w * 16 + t]; }
        }
        sRowMax[t] = mx; sRowIdx[t] = mi;
    }
    __syncthreads();

    // ---------------- Phase 3: exp + row sums (in LDS) ----------------------
    {
        const int row = t >> 4, c0 = t & 15;
        const float mx = sRowMax[row];
        float lsum = 0.f;
        for (int n = c0; n < K_CB; n += 16) {
            float e = __expf((float)sP[row * K_CB + n] - mx);
            sP[row * K_CB + n] = (__bf16)e;
            lsum += e;
        }
        atomicAdd(&sRowSum[row], lsum);
    }
    __syncthreads();
    if (t < 16) sRowInv[t] = 1.0f / sRowSum[t];
    __syncthreads();

    // ---------------- Phase 4: normalize + stream encodings to HBM ----------
    {
        const int row = t >> 4, c0 = t & 15;
        const float inv = sRowInv[row];
        float* dst = enc_out + (pix0 + row) * (size_t)K_CB;
        for (int n = c0; n < K_CB; n += 16) {
            float p = (float)sP[row * K_CB + n] * inv;
            sP[row * K_CB + n] = (__bf16)p;
            dst[n] = p;                       // single pass: NT-ish streaming store
        }
    }
    __syncthreads();

    // ---------------- Phase 5: GEMM2  quantized = probs @ codebook ----------
    for (int tile = wave; tile < D_DIM / 16; tile += NWAVES) {
        const int d0 = tile * 16;
        const int m  = lane & 15;
        const int bo = (lane >> 4) ? 16 : 0;
        const __bf16* bcol = &cbT16[(size_t)(d0 + ncol) * K_CB + ((lane >> 4) << 4)];
        v8f acc = {};
        for (int k0 = 0; k0 < K_CB; k0 += 32) {
            const __bf16* ap = &sP[m * K_CB + k0 + bo];
            v8bf lo = *(const v8bf*)(ap);
            v8bf hi = *(const v8bf*)(ap + 8);
            v16bf a;
            #pragma unroll
            for (int i = 0; i < 8; ++i) { a[i] = lo[i]; a[i + 8] = hi[i]; }
            v16bf b = *(const v16bf*)(bcol + k0);
            acc = __builtin_amdgcn_wmma_f32_16x16x32_bf16(
                      false, a, false, b, (short)0, acc, false, false);
        }
        #pragma unroll
        for (int r = 0; r < 8; ++r)
            q_out[(pix0 + r + hi8) * D_DIM + d0 + ncol] = acc[r];
    }

    // ---------------- Phase 6: argmin indices --------------------------------
    if (t < 16) idx_out[pix0 + t] = sRowIdx[t];
}

// ---------------------------------------------------------------------------
// Host launcher
// ---------------------------------------------------------------------------
extern "C" void kernel_launch(void* const* d_in, const int* in_sizes, int n_in,
                              void* d_out, int out_size, void* d_ws, size_t ws_size,
                              hipStream_t stream) {
    (void)in_sizes; (void)n_in; (void)out_size; (void)ws_size;
    const float* x        = (const float*)d_in[0];   // [16,32,32,256]
    const float* codebook = (const float*)d_in[1];   // [8192,256]
    const float* noise    = (const float*)d_in[2];   // [16,32,32,8192]

    // d_out: quantized | encodings | indices (return order)
    float* q_out   = (float*)d_out;
    float* enc_out = q_out + (size_t)N_PIX * D_DIM;
    int*   idx_out = (int*)(enc_out + (size_t)N_PIX * K_CB);

    // workspace: cb16 (4MB) | cbT16 (4MB) | c2 (32KB)
    __bf16* cb16  = (__bf16*)d_ws;
    __bf16* cbT16 = cb16 + (size_t)K_CB * D_DIM;
    float*  c2    = (float*)(cbT16 + (size_t)K_CB * D_DIM);

    vq_prep<<<K_CB, 256, 0, stream>>>(codebook, cb16, cbT16, c2);

    static int smem_set = 0;  // attribute set is idempotent & capture-safe
    if (!smem_set) {
        hipFuncSetAttribute((const void*)vq_main,
                            hipFuncAttributeMaxDynamicSharedMemorySize, SMEM_BYTES);
        smem_set = 1;
    }
    vq_main<<<N_PIX / ROWS, 256, SMEM_BYTES, stream>>>(
        x, noise, cb16, cbT16, c2, q_out, enc_out, idx_out);
}
